// ConvMlp4d_23441931501668
// MI455X (gfx1250) — matrix operand order repaired and checked by reading, not compiled, not run on hardware
//
#include <hip/hip_runtime.h>
#include <hip/hip_bf16.h>
#include <stddef.h>

// ---------------------------------------------------------------------------
// Types matching CDNA5 WMMA builtin signatures
// ---------------------------------------------------------------------------
typedef __attribute__((ext_vector_type(16))) __bf16      bf16x16;
typedef __attribute__((ext_vector_type(8)))  float       f32x8;
typedef __attribute__((ext_vector_type(4)))  unsigned int u32x4;

union Frag { u32x4 q[2]; bf16x16 v; };

// Round-to-nearest-even f32 -> bf16
static __device__ __forceinline__ __bf16 f2bf(float f) {
    union { float f; unsigned u; } in; in.f = f;
    unsigned u = in.u;
    u += 0x7FFFu + ((u >> 16) & 1u);
    union { unsigned short s; __bf16 b; } out; out.s = (unsigned short)(u >> 16);
    return out.b;
}

// ---------------------------------------------------------------------------
// Fixed shapes: x (2,64,4,16,32,32) -> 131072 positions (power-of-two dims)
//   position p bits: w[4:0] v[9:5] u[13:10] t[15:14] n[16]
// ---------------------------------------------------------------------------
#define NPOS   131072
#define SPB    65536
#define TAPS   81

// Workspace layout (bytes)
#define XP_OFF   0u
#define W1T_OFF  16777216u
#define W2T_OFF  (16777216u + 1327104u)
#define HID_OFF  (16777216u + 2u * 1327104u)
#define ZP_OFF   (HID_OFF + 33554432u)         // 512B zero page

// ---------------------------------------------------------------------------
// Prepass kernels
// ---------------------------------------------------------------------------
__global__ void __launch_bounds__(128) init_zpage_kernel(unsigned int* z) {
    z[threadIdx.x] = 0u;                       // 128 * 4B = 512B
}

__global__ void __launch_bounds__(256) cvt_x_kernel(const float* __restrict__ x,
                                                    __bf16* __restrict__ xp) {
    const int i = blockIdx.x * 256 + threadIdx.x;   // total 2*64*65536
    const int c   = i & 63;
    const int pos = i >> 6;
    const int nb  = pos >> 16;
    const int s   = pos & (SPB - 1);
    xp[i] = f2bf(x[((size_t)(nb * 64 + c)) * SPB + s]);
}

// w (O,I,3,3,3,3) f32 -> wt [tap][O][I] bf16
__global__ void __launch_bounds__(256) cvt_w_kernel(const float* __restrict__ w,
                                                    __bf16* __restrict__ wt,
                                                    int O, int I, int total) {
    const int i = blockIdx.x * 256 + threadIdx.x;
    if (i >= total) return;
    const int ii   = i % I;
    const int rest = i / I;
    const int o    = rest % O;
    const int tap  = rest / O;
    wt[i] = f2bf(w[((size_t)(o * I + ii)) * TAPS + tap]);
}

// ---------------------------------------------------------------------------
// Implicit-GEMM 4D conv, V_WMMA_F32_16X16X32_BF16.
// Wave = 4 M-tiles x 2 N-tiles (64x32 output): per (tap,kb) load 4 A + 2 B
// fragments, then run 8 back-to-back WMMAs (one wait per 8 matrix ops).
// OOB taps read a zeroed page via pointer select.
// ---------------------------------------------------------------------------
template <int CIN, int COUT, bool FUSE>
__global__ void __launch_bounds__(256, 1)
conv4d_wmma_kernel(const __bf16* __restrict__ xin,
                   const __bf16* __restrict__ wt,
                   const float*  __restrict__ bias,
                   const __bf16* __restrict__ zpage,
                   __bf16*       __restrict__ hout,
                   float*        __restrict__ fout) {
    constexpr int NT2  = COUT / 32;           // N tile-pairs
    constexpr int KB   = CIN / 32;            // K blocks per tap
    constexpr int SIXT = 16 * CIN;            // element offset to 2nd B tile
    const int lane  = threadIdx.x & 31;
    const int wave  = threadIdx.x >> 5;
    const int nt2   = blockIdx.x % NT2;       // all 8 waves share the N pair
    const int mgrp  = (blockIdx.x / NT2) * 8 + wave;
    const int row   = lane & 15;
    const int hi    = lane >> 4;
    const int och   = nt2 * 32 + row;         // first N tile's column
    // second N tile's column = och + 16

    f32x8 acc[4][2];
#pragma unroll
    for (int i = 0; i < 4; ++i)
#pragma unroll
        for (int n = 0; n < 2; ++n)
#pragma unroll
            for (int j = 0; j < 8; ++j) acc[i][n][j] = 0.0f;

    // This lane's four M rows: coords + base pointers (position m == its own
    // flat spatial index, all dims power-of-two).
    int w0[4], v0[4], u0[4], t0[4];
    const __bf16* base[4];
#pragma unroll
    for (int mi = 0; mi < 4; ++mi) {
        const int m = mgrp * 64 + mi * 16 + row;
        w0[mi] = m & 31;
        v0[mi] = (m >> 5) & 31;
        u0[mi] = (m >> 10) & 15;
        t0[mi] = (m >> 14) & 3;
        base[mi] = xin + (size_t)m * CIN;
    }

    const __bf16* wtap = wt + (size_t)och * CIN;      // tap 0 row, N tile 0
    const ptrdiff_t stepW = (ptrdiff_t)COUT * CIN;    // per-tap advance

#pragma clang loop unroll(disable)
    for (int dt = -1; dt <= 1; ++dt) {
#pragma clang loop unroll(disable)
        for (int du = -1; du <= 1; ++du) {
#pragma clang loop unroll(disable)
            for (int dv = -1; dv <= 1; ++dv) {
                for (int dw = -1; dw <= 1; ++dw) {
                    __builtin_prefetch(wtap + stepW, 0, 3);

                    // Uniform tap delta (elements); per-tile pointer = base+d
                    const ptrdiff_t d =
                        (ptrdiff_t)(dt * 16384 + du * 1024 + dv * 32 + dw) * CIN;

                    const __bf16* s0[4];
#pragma unroll
                    for (int mi = 0; mi < 4; ++mi) {
                        const bool ok = ((unsigned)(w0[mi] + dw) < 32u) &
                                        ((unsigned)(v0[mi] + dv) < 32u) &
                                        ((unsigned)(u0[mi] + du) < 16u) &
                                        ((unsigned)(t0[mi] + dt) < 4u);
                        const __bf16* p = base[mi] + d;
                        s0[mi] = ok ? p : zpage;
                    }

#pragma unroll
                    for (int k = 0; k < KB; ++k) {
                        const int kb = k * 32;
                        // B fragments (32B-aligned): two N tiles
                        bf16x16 fbA = *(const bf16x16*)(wtap + kb + hi * 16);
                        bf16x16 fbB = *(const bf16x16*)(wtap + SIXT + kb + hi * 16);
                        // A fragments: 4 M tiles
                        Frag fa[4];
#pragma unroll
                        for (int mi = 0; mi < 4; ++mi) {
                            fa[mi].q[0] = *(const u32x4*)(s0[mi] + kb + hi * 16);
                            fa[mi].q[1] = *(const u32x4*)(s0[mi] + kb + hi * 16 + 8);
                        }
                        // 8 back-to-back matrix ops
#pragma unroll
                        for (int mi = 0; mi < 4; ++mi) {
                            acc[mi][0] = __builtin_amdgcn_wmma_f32_16x16x32_bf16(
                                false, fa[mi].v, false, fbA, (short)0,
                                acc[mi][0], false, false);
                            acc[mi][1] = __builtin_amdgcn_wmma_f32_16x16x32_bf16(
                                false, fa[mi].v, false, fbB, (short)0,
                                acc[mi][1], false, false);
                        }
                    }
                    wtap += stepW;
                }
            }
        }
    }

    // Epilogue. D layout: N = lane&15, M = hi*8 + r. Two channels per lane.
    const float bv0 = bias[och];
    const float bv1 = bias[och + 16];
#pragma unroll
    for (int mi = 0; mi < 4; ++mi) {
#pragma unroll
        for (int r = 0; r < 8; ++r) {
            float y0 = acc[mi][0][r] + bv0;
            float y1 = acc[mi][1][r] + bv1;
            const int p = mgrp * 64 + mi * 16 + hi * 8 + r;
            if constexpr (FUSE) {
                y0 = fmaxf(y0, 0.0f);
                y1 = fmaxf(y1, 0.0f);
                hout[(size_t)p * COUT + och]      = f2bf(y0);
                hout[(size_t)p * COUT + och + 16] = f2bf(y1);
            } else {
                const int tt = (p >> 14) & 3;
                const int uu = (p >> 10) & 15;
                const int nbb = p >> 16;
                const size_t sp = (p & 1023) + ((size_t)tt * 16 + uu) * 1024;
                fout[((size_t)nbb * COUT + och)      * 65536 + sp] = y0;
                fout[((size_t)nbb * COUT + och + 16) * 65536 + sp] = y1;
            }
        }
    }
}

// ---------------------------------------------------------------------------
// Launcher. Inputs: x, w1, b1, w2, b2 (f32). Output f32 (2,64,4,16,32,32).
// ---------------------------------------------------------------------------
extern "C" void kernel_launch(void* const* d_in, const int* in_sizes, int n_in,
                              void* d_out, int out_size, void* d_ws, size_t ws_size,
                              hipStream_t stream) {
    const float* x  = (const float*)d_in[0];
    const float* w1 = (const float*)d_in[1];
    const float* b1 = (const float*)d_in[2];
    const float* w2 = (const float*)d_in[3];
    const float* b2 = (const float*)d_in[4];
    float* out = (float*)d_out;

    char* ws = (char*)d_ws;
    __bf16* xp    = (__bf16*)(ws + XP_OFF);
    __bf16* w1t   = (__bf16*)(ws + W1T_OFF);
    __bf16* w2t   = (__bf16*)(ws + W2T_OFF);
    __bf16* hid   = (__bf16*)(ws + HID_OFF);
    __bf16* zpage = (__bf16*)(ws + ZP_OFF);

    init_zpage_kernel<<<1, 128, 0, stream>>>((unsigned int*)zpage);
    cvt_x_kernel<<<(NPOS * 64) / 256, 256, 0, stream>>>(x, xp);
    const int wtot = TAPS * 128 * 64;
    cvt_w_kernel<<<(wtot + 255) / 256, 256, 0, stream>>>(w1, w1t, 128, 64, wtot);
    cvt_w_kernel<<<(wtot + 255) / 256, 256, 0, stream>>>(w2, w2t, 64, 128, wtot);

    // conv1: 64 -> 128, bias+ReLU -> bf16 channel-last hidden
    //   blocks = NT2(4) * (2048 mgroups / 8 waves) = 1024
    conv4d_wmma_kernel<64, 128, true><<<1024, 256, 0, stream>>>(
        xp, w1t, b1, zpage, hid, nullptr);

    // conv2: 128 -> 64, bias -> f32 NCTUVW output
    //   blocks = NT2(2) * (2048 / 8) = 512
    conv4d_wmma_kernel<128, 64, false><<<512, 256, 0, stream>>>(
        hid, w2t, b2, zpage, nullptr, out);
}
